// MOELayer_6536940225052
// MI455X (gfx1250) — matrix operand order repaired and checked by reading, not compiled, not run on hardware
//
#include <hip/hip_runtime.h>
#include <cstdint>

// ---------------------------------------------------------------------------
// MoE Top-2 layer for MI455X (gfx1250, wave32, WMMA).
// S=8192 tokens, M=1024 model dim, E=8 experts, V=4096 hidden, CAP=2048.
// Expert GEMMs (~275 GFLOP) use v_wmma_f32_16x16x32_bf16 (bf16 in, f32 acc):
// 16 KFLOP/instr vs 2 KFLOP for the f32 K=4 WMMA -> ~8x matrix throughput.
// The 192MB L2 holds each expert's working set, so fp32->bf16 conversion
// re-reads are L2-resident.  GEMMs are double-buffered: next chunk's global
// loads are issued before the WMMAs; the bf16 A operand of GEMM2 is copied
// with gfx1250 global_load_async_to_lds_b128 (ASYNCcnt) straight into LDS.
// ---------------------------------------------------------------------------

#define S_TOK   8192
#define M_DIM   1024
#define E_EXP   8
#define V_DIM   4096
#define CAPACITY 2048            // 2*ceil(S/E); power of two
#define EC      (E_EXP * CAPACITY)

typedef __attribute__((ext_vector_type(16))) __bf16 v16bf;
typedef __attribute__((ext_vector_type(8)))  float  v8f;

// ---------------------------------------------------------------------------
// Kernel 0: zero the dispatch buffer (float4 stores, exact coverage)
// ---------------------------------------------------------------------------
__global__ __launch_bounds__(256) void zero4_kernel(float4* __restrict__ p, int n4) {
    int i = blockIdx.x * 256 + threadIdx.x;
    if (i < n4) p[i] = make_float4(0.f, 0.f, 0.f, 0.f);
}

// ---------------------------------------------------------------------------
// Kernel 1: gating logits + softmax + top2.  One wave per token.
// ---------------------------------------------------------------------------
__global__ __launch_bounds__(256)
void gate_kernel(const float* __restrict__ x, const float* __restrict__ wg,
                 float* __restrict__ gates, int* __restrict__ idx1a,
                 int* __restrict__ idx2a, float* __restrict__ g1a,
                 float* __restrict__ g2a) {
    const int wave = threadIdx.x >> 5;
    const int lane = threadIdx.x & 31;
    const int tok  = blockIdx.x * 8 + wave;
    const float* xr = x + (size_t)tok * M_DIM;

    float s[E_EXP];
#pragma unroll
    for (int e = 0; e < E_EXP; ++e) s[e] = 0.f;

    for (int d = lane; d < M_DIM; d += 32) {
        const float xv = xr[d];
#pragma unroll
        for (int e = 0; e < E_EXP; ++e) s[e] += xv * wg[e * M_DIM + d];
    }
#pragma unroll
    for (int e = 0; e < E_EXP; ++e)
        for (int off = 16; off > 0; off >>= 1) s[e] += __shfl_xor(s[e], off, 32);

    // softmax (all lanes redundantly)
    float mx = s[0];
#pragma unroll
    for (int e = 1; e < E_EXP; ++e) mx = fmaxf(mx, s[e]);
    float p[E_EXP], sum = 0.f;
#pragma unroll
    for (int e = 0; e < E_EXP; ++e) { p[e] = expf(s[e] - mx); sum += p[e]; }
    const float inv = 1.f / sum;
#pragma unroll
    for (int e = 0; e < E_EXP; ++e) p[e] *= inv;

    // top-1 / top-2 with first-index tie break (matches jax top_k)
    int i1 = 0; float b1 = s[0];
#pragma unroll
    for (int e = 1; e < E_EXP; ++e) if (s[e] > b1) { b1 = s[e]; i1 = e; }
    int i2 = -1; float b2 = -3.4e38f;
#pragma unroll
    for (int e = 0; e < E_EXP; ++e)
        if (e != i1 && s[e] > b2) { b2 = s[e]; i2 = e; }

    if (lane < E_EXP) gates[(size_t)tok * E_EXP + lane] = p[lane];
    if (lane == 0) {
        idx1a[tok] = i1; idx2a[tok] = i2;
        g1a[tok] = p[i1]; g2a[tok] = p[i2];
    }
}

// ---------------------------------------------------------------------------
// Kernel 2: single-block ordered scan -> slot positions, flat indices, l_aux.
// 512 threads x 16 tokens each.  Token order must match the reference cumsum.
// ---------------------------------------------------------------------------
#define SCAN_T  512
#define TOK_PER 16

__global__ __launch_bounds__(SCAN_T)
void scan_kernel(const float* __restrict__ gates, const int* __restrict__ idx1a,
                 const int* __restrict__ idx2a, int* __restrict__ flat1,
                 int* __restrict__ flat2, float* __restrict__ laux_dst) {
    __shared__ int sc[SCAN_T * E_EXP];
    __shared__ int tot1[E_EXP];
    const int t = threadIdx.x;
    const int base = t * TOK_PER;

    int i1loc[TOK_PER], i2loc[TOK_PER];
    int c1[E_EXP], c2[E_EXP];
#pragma unroll
    for (int e = 0; e < E_EXP; ++e) { c1[e] = 0; c2[e] = 0; }
    for (int i = 0; i < TOK_PER; ++i) {
        i1loc[i] = idx1a[base + i]; c1[i1loc[i]]++;
        i2loc[i] = idx2a[base + i]; c2[i2loc[i]]++;
    }

    // inclusive Hillis-Steele scan of c1
#pragma unroll
    for (int e = 0; e < E_EXP; ++e) sc[t * E_EXP + e] = c1[e];
    __syncthreads();
    for (int d = 1; d < SCAN_T; d <<= 1) {
        int add[E_EXP];
        if (t >= d)
            for (int e = 0; e < E_EXP; ++e) add[e] = sc[(t - d) * E_EXP + e];
        __syncthreads();
        if (t >= d)
            for (int e = 0; e < E_EXP; ++e) sc[t * E_EXP + e] += add[e];
        __syncthreads();
    }
    int ex1[E_EXP];
#pragma unroll
    for (int e = 0; e < E_EXP; ++e) ex1[e] = sc[t * E_EXP + e] - c1[e];
    if (t == SCAN_T - 1)
        for (int e = 0; e < E_EXP; ++e) tot1[e] = sc[t * E_EXP + e];
    __syncthreads();

    // scan of c2 (reuse sc)
#pragma unroll
    for (int e = 0; e < E_EXP; ++e) sc[t * E_EXP + e] = c2[e];
    __syncthreads();
    for (int d = 1; d < SCAN_T; d <<= 1) {
        int add[E_EXP];
        if (t >= d)
            for (int e = 0; e < E_EXP; ++e) add[e] = sc[(t - d) * E_EXP + e];
        __syncthreads();
        if (t >= d)
            for (int e = 0; e < E_EXP; ++e) sc[t * E_EXP + e] += add[e];
        __syncthreads();
    }
    int ex2[E_EXP];
#pragma unroll
    for (int e = 0; e < E_EXP; ++e) ex2[e] = sc[t * E_EXP + e] - c2[e];
    __syncthreads();

    int t1[E_EXP];
#pragma unroll
    for (int e = 0; e < E_EXP; ++e) t1[e] = tot1[e];

    // assign positions in token order; dropped tokens alias slot 0 (reference)
    int r1[E_EXP], r2[E_EXP];
#pragma unroll
    for (int e = 0; e < E_EXP; ++e) { r1[e] = 0; r2[e] = 0; }
    for (int i = 0; i < TOK_PER; ++i) {
        const int tok = base + i;
        const int e1 = i1loc[i];
        const int pos1 = ex1[e1] + r1[e1]; r1[e1]++;
        flat1[tok] = e1 * CAPACITY + (pos1 < CAPACITY ? pos1 : 0);
        const int e2 = i2loc[i];
        const int pos2 = ex2[e2] + r2[e2] + t1[e2]; r2[e2]++;
        flat2[tok] = e2 * CAPACITY + (pos2 < CAPACITY ? pos2 : 0);
    }

    // l_aux = sum_e me[e]*ce[e] * E/S^2 ;  ce = min(count1, CAP)
    float gm[E_EXP];
#pragma unroll
    for (int e = 0; e < E_EXP; ++e) gm[e] = 0.f;
    for (int i = 0; i < TOK_PER; ++i)
#pragma unroll
        for (int e = 0; e < E_EXP; ++e)
            gm[e] += gates[(size_t)(base + i) * E_EXP + e];
    float* sf = (float*)sc;
#pragma unroll
    for (int e = 0; e < E_EXP; ++e) sf[t * E_EXP + e] = gm[e];
    __syncthreads();
    for (int d = SCAN_T / 2; d > 0; d >>= 1) {
        if (t < d)
            for (int e = 0; e < E_EXP; ++e)
                sf[t * E_EXP + e] += sf[(t + d) * E_EXP + e];
        __syncthreads();
    }
    if (t == 0) {
        float l = 0.f;
        for (int e = 0; e < E_EXP; ++e) {
            const float ce = (float)(t1[e] < CAPACITY ? t1[e] : CAPACITY);
            l += sf[e] * ce;
        }
        *laux_dst = l * ((float)E_EXP / ((float)S_TOK * (float)S_TOK));
    }
}

// ---------------------------------------------------------------------------
// Kernel 3: dispatch scatter.  Slot 0 of an expert can receive multiple tokens
// (the legit slot-0 token plus all capacity-dropped tokens) -> atomicAdd there;
// every other slot has a unique writer -> plain float4 store.
// ---------------------------------------------------------------------------
__global__ __launch_bounds__(256)
void scatter_kernel(const float* __restrict__ x, const int* __restrict__ flat1,
                    const int* __restrict__ flat2, float* __restrict__ disp) {
    const int tok = blockIdx.x;
    const int col = threadIdx.x * 4;
    const float4 xv = *(const float4*)(x + (size_t)tok * M_DIM + col);

    const int f1 = flat1[tok];
    float* d1 = disp + (size_t)f1 * M_DIM + col;
    if ((f1 & (CAPACITY - 1)) == 0) {
        atomicAdd(d1 + 0, xv.x); atomicAdd(d1 + 1, xv.y);
        atomicAdd(d1 + 2, xv.z); atomicAdd(d1 + 3, xv.w);
    } else {
        *(float4*)d1 = xv;
    }
    const int f2 = flat2[tok];
    float* d2 = disp + (size_t)f2 * M_DIM + col;
    if ((f2 & (CAPACITY - 1)) == 0) {
        atomicAdd(d2 + 0, xv.x); atomicAdd(d2 + 1, xv.y);
        atomicAdd(d2 + 2, xv.z); atomicAdd(d2 + 3, xv.w);
    } else {
        *(float4*)d2 = xv;
    }
}

// ---------------------------------------------------------------------------
// Kernel 4/5: tiled WMMA GEMM.  Per expert:  Out[CAP x N] = A[CAP x K] * B[K x N]
// + bias (+ReLU).  128x128 C-tile per 256-thread block (8 waves, 4x2 layout),
// each wave computes 32x64 = eight v_wmma_f32_16x16x32_bf16 per 32-wide K chunk.
// Double-buffered LDS (80B pitch, conflict-free 16B frag loads):
//   - fp32 operands: global->VGPR prefetch issued before the WMMAs, converted
//     to bf16 and stored to the other buffer afterwards.
//   - bf16 A operand (GEMM2): gfx1250 global_load_async_to_lds_b128 copies
//     straight into LDS (ASYNCcnt); published with s_wait_asynccnt + barrier
//     (__syncthreads only covers DScnt).
// Fragment layouts (contiguous k order, same for A and B):
//   A lane(h=lane>>4, r=lane&15): elems[0..15] = K[h*16 .. h*16+15] of row r
//   B lane(h, c): elems[0..15] = K[h*16 .. h*16+15] of column c
//   C/D: vgpr g, lane -> (row = h*8+g, col = c)
// ---------------------------------------------------------------------------
#define BT  128
#define KT  32
#define LDT 40

union FragBF { v16bf v; float4 q[2]; };

template <bool A_BF16, bool RELU, bool OUT_BF16>
__global__ __launch_bounds__(256)
void moe_gemm(const void* __restrict__ Ap, const float* __restrict__ Bp,
              const float* __restrict__ biasp, void* __restrict__ Op,
              int K, int N) {
    const int e = blockIdx.z;
    const size_t aStride = (size_t)CAPACITY * K;
    const size_t bStride = (size_t)K * N;
    const size_t oStride = (size_t)CAPACITY * N;
    const float* Bw   = Bp + (size_t)e * bStride;
    const float* bias = biasp + (size_t)e * N;
    const float*  Af = A_BF16 ? nullptr : ((const float*)Ap + (size_t)e * aStride);
    const __bf16* Ah = A_BF16 ? ((const __bf16*)Ap + (size_t)e * aStride) : nullptr;

    __shared__ __align__(16) __bf16 As[2][BT * LDT];
    __shared__ __align__(16) __bf16 Bs[2][BT * LDT];

    const int tid  = threadIdx.x;
    const int wave = tid >> 5, lane = tid & 31;
    const int wm = wave & 3, wn = wave >> 2;       // 4 M-waves x 2 N-waves
    const int h = lane >> 4, c = lane & 15;
    const int tileM = blockIdx.x * BT;
    const int tileN = blockIdx.y * BT;

    // --- prefetch / staging helpers -------------------------------------
    float4 pa[4];   // fp32 A staging (GEMM1)
    float4 pb[4];   // fp32 B staging (both GEMMs)

    auto issueA_async = [&](int kk, int buf) {   // bf16 A: direct to LDS
#pragma unroll
        for (int it = 0; it < 2; ++it) {
            const int slot = tid + it * 256;     // 512 x 8-bf16 chunks
            const int row = slot >> 2, cg = (slot & 3) << 3;
            const unsigned lds =
                (unsigned)(uintptr_t)(&As[buf][row * LDT + cg]);
            const unsigned long long ga = (unsigned long long)
                (Ah + (size_t)(tileM + row) * K + kk + cg);
            asm volatile("global_load_async_to_lds_b128 %0, %1, off"
                         :: "v"(lds), "v"(ga) : "memory");
        }
    };
    auto loadA_f32 = [&](int kk) {
#pragma unroll
        for (int it = 0; it < 4; ++it) {
            const int slot = tid + it * 256;     // 1024 x 4-float chunks
            const int row = slot >> 3, cg = (slot & 7) << 2;
            pa[it] = *(const float4*)(Af + (size_t)(tileM + row) * K + kk + cg);
        }
    };
    auto storeA_f32 = [&](int buf) {
#pragma unroll
        for (int it = 0; it < 4; ++it) {
            const int slot = tid + it * 256;
            const int row = slot >> 3, cg = (slot & 7) << 2;
            __bf16* d = &As[buf][row * LDT + cg];
            d[0] = (__bf16)pa[it].x; d[1] = (__bf16)pa[it].y;
            d[2] = (__bf16)pa[it].z; d[3] = (__bf16)pa[it].w;
        }
    };
    auto loadB = [&](int kk) {
#pragma unroll
        for (int it = 0; it < 4; ++it) {
            const int slot = tid + it * 256;     // 32 k x 32 n-groups
            const int k = slot >> 5, ng = (slot & 31) << 2;
            pb[it] = *(const float4*)(Bw + (size_t)(kk + k) * N + tileN + ng);
        }
    };
    auto storeB = [&](int buf) {                 // transpose to Bs[n][k]
#pragma unroll
        for (int it = 0; it < 4; ++it) {
            const int slot = tid + it * 256;
            const int k = slot >> 5, ng = (slot & 31) << 2;
            Bs[buf][(ng + 0) * LDT + k] = (__bf16)pb[it].x;
            Bs[buf][(ng + 1) * LDT + k] = (__bf16)pb[it].y;
            Bs[buf][(ng + 2) * LDT + k] = (__bf16)pb[it].z;
            Bs[buf][(ng + 3) * LDT + k] = (__bf16)pb[it].w;
        }
    };

    v8f acc[2][4];
#pragma unroll
    for (int mt = 0; mt < 2; ++mt)
#pragma unroll
        for (int nt = 0; nt < 4; ++nt)
#pragma unroll
            for (int g = 0; g < 8; ++g) acc[mt][nt][g] = 0.f;

    // --- prologue: fill buffer 0 ----------------------------------------
    if (A_BF16) {
        issueA_async(0, 0);
    } else {
        loadA_f32(0);
        storeA_f32(0);
    }
    loadB(0);
    storeB(0);
    if (A_BF16) asm volatile("s_wait_asynccnt 0x0" ::: "memory");
    __syncthreads();

    int cur = 0;
    for (int k0 = 0; k0 < K; k0 += KT) {
        const int nxt = cur ^ 1;
        const bool hasNext = (k0 + KT) < K;

        // issue next chunk's global traffic before the matrix ops
        if (hasNext) {
            if (A_BF16) issueA_async(k0 + KT, nxt);
            else        loadA_f32(k0 + KT);
            loadB(k0 + KT);
        }

        // ---- fragments + WMMA on current buffer ----
        FragBF fa[2], fb[4];
#pragma unroll
        for (int mt = 0; mt < 2; ++mt) {
            const int row = wm * 32 + mt * 16 + c;
            fa[mt].q[0] = *(const float4*)(&As[cur][row * LDT + h * 16]);
            fa[mt].q[1] = *(const float4*)(&As[cur][row * LDT + h * 16 + 8]);
        }
#pragma unroll
        for (int nt = 0; nt < 4; ++nt) {
            const int nr = wn * 64 + nt * 16 + c;
            fb[nt].q[0] = *(const float4*)(&Bs[cur][nr * LDT + h * 16]);
            fb[nt].q[1] = *(const float4*)(&Bs[cur][nr * LDT + h * 16 + 8]);
        }
#pragma unroll
        for (int mt = 0; mt < 2; ++mt)
#pragma unroll
            for (int nt = 0; nt < 4; ++nt)
                acc[mt][nt] = __builtin_amdgcn_wmma_f32_16x16x32_bf16(
                    false, fa[mt].v, false, fb[nt].v,
                    (short)0, acc[mt][nt], false, false);

        // ---- publish next buffer ----
        if (hasNext) {
            if (!A_BF16) storeA_f32(nxt);
            storeB(nxt);
            if (A_BF16) asm volatile("s_wait_asynccnt 0x0" ::: "memory");
            __syncthreads();   // waits DScnt, then barrier
            cur = nxt;
        }
    }

    // ---- epilogue: bias (+ReLU), store bf16 or f32 ----
#pragma unroll
    for (int mt = 0; mt < 2; ++mt)
#pragma unroll
        for (int nt = 0; nt < 4; ++nt) {
            const int col = tileN + wn * 64 + nt * 16 + c;
            const float bv = bias[col];
#pragma unroll
            for (int g = 0; g < 8; ++g) {
                const int row = tileM + wm * 32 + mt * 16 + h * 8 + g;
                float val = acc[mt][nt][g] + bv;
                if (RELU) val = fmaxf(val, 0.f);
                const size_t oi = (size_t)e * oStride + (size_t)row * N + col;
                if (OUT_BF16) ((__bf16*)Op)[oi] = (__bf16)val;
                else          ((float*)Op)[oi]  = val;
            }
        }
}

// ---------------------------------------------------------------------------
// Kernel 6: combine  out = g1*eo[flat1] + g2*eo[flat2]
// ---------------------------------------------------------------------------
__global__ __launch_bounds__(256)
void combine_kernel(const float* __restrict__ eo, const int* __restrict__ flat1,
                    const int* __restrict__ flat2, const float* __restrict__ g1a,
                    const float* __restrict__ g2a, float* __restrict__ out) {
    const int tok = blockIdx.x;
    const int col = threadIdx.x * 4;
    const float g1 = g1a[tok], g2 = g2a[tok];
    const int f1 = flat1[tok], f2 = flat2[tok];
    const float4 a = *(const float4*)(eo + (size_t)f1 * M_DIM + col);
    const float4 b = *(const float4*)(eo + (size_t)f2 * M_DIM + col);
    float4 o;
    o.x = g1 * a.x + g2 * b.x;
    o.y = g1 * a.y + g2 * b.y;
    o.z = g1 * a.z + g2 * b.z;
    o.w = g1 * a.w + g2 * b.w;
    *(float4*)(out + (size_t)tok * M_DIM + col) = o;
}

// ---------------------------------------------------------------------------
// Host launch
// ---------------------------------------------------------------------------
extern "C" void kernel_launch(void* const* d_in, const int* in_sizes, int n_in,
                              void* d_out, int out_size, void* d_ws, size_t ws_size,
                              hipStream_t stream) {
    (void)in_sizes; (void)n_in; (void)out_size; (void)ws_size;
    const float* x     = (const float*)d_in[0];
    const float* wg    = (const float*)d_in[1];
    const float* fc1_w = (const float*)d_in[2];
    const float* fc1_b = (const float*)d_in[3];
    const float* fc2_w = (const float*)d_in[4];
    const float* fc2_b = (const float*)d_in[5];
    float* out = (float*)d_out;

    // workspace layout
    char* ws = (char*)d_ws;
    float*  disp  = (float*)ws;                                  // 64 MB (EO reuses it)
    __bf16* H     = (__bf16*)(ws + ((size_t)64 << 20));          // 128 MB
    float*  gates = (float*)(ws + ((size_t)192 << 20));          // S*E
    float*  g1    = gates + (size_t)S_TOK * E_EXP;
    float*  g2    = g1 + S_TOK;
    int*    flat1 = (int*)(g2 + S_TOK);
    int*    flat2 = flat1 + S_TOK;
    int*    idx1  = flat2 + S_TOK;
    int*    idx2  = idx1 + S_TOK;

    // 0) zero dispatch buffer (EC*M floats)
    const int n4 = (EC * M_DIM) / 4;
    zero4_kernel<<<n4 / 256, 256, 0, stream>>>((float4*)disp, n4);

    // 1) gating
    gate_kernel<<<S_TOK / 8, 256, 0, stream>>>(x, wg, gates, idx1, idx2, g1, g2);

    // 2) ordered scan -> flat indices + l_aux (written to d_out[S*M])
    scan_kernel<<<1, SCAN_T, 0, stream>>>(gates, idx1, idx2, flat1, flat2,
                                          out + (size_t)S_TOK * M_DIM);

    // 3) dispatch scatter
    scatter_kernel<<<S_TOK, 256, 0, stream>>>(x, flat1, flat2, disp);

    // 4) GEMM1: H = relu(D @ W1 + b1)   (fp32 A, bf16 out)
    moe_gemm<false, true, true><<<dim3(CAPACITY / BT, V_DIM / BT, E_EXP),
                                  256, 0, stream>>>(disp, fc1_w, fc1_b, H,
                                                    M_DIM, V_DIM);

    // 5) GEMM2: EO = H @ W2 + b2        (bf16 A via async LDS copy, fp32 out)
    moe_gemm<true, false, false><<<dim3(CAPACITY / BT, M_DIM / BT, E_EXP),
                                   256, 0, stream>>>(H, fc2_w, fc2_b, disp,
                                                     V_DIM, M_DIM);

    // 6) combine
    combine_kernel<<<S_TOK, 256, 0, stream>>>(disp, flat1, flat2, g1, g2, out);
}
